// MultiHeadedAttention_28252294873267
// MI455X (gfx1250) — matrix operand order repaired and checked by reading, not compiled, not run on hardware
//
#include <hip/hip_runtime.h>

// MI455X / gfx1250, wave32. bf16 WMMA (v_wmma_f32_16x16x32_bf16) everywhere.
typedef __attribute__((ext_vector_type(16))) __bf16 v16bf;
typedef __attribute__((ext_vector_type(8)))  __bf16 v8bf;
typedef __attribute__((ext_vector_type(4)))  __bf16 v4bf;
typedef __attribute__((ext_vector_type(8)))  float  v8f;

#define NB   2
#define NS   2048
#define ND   1024
#define NH   16
#define NHS  64
#define ATTN_SCALE 0.125f   // 1/sqrt(64)

static __device__ __forceinline__ unsigned laneid() { return threadIdx.x & 31u; }

// ---- WMMA fragment loaders (ISA 7.12.2 wave32 layouts) ----------------------
// A (16x32 bf16, MxK): lane<16 -> row=lane, K in {k0..k0+7, k0+16..k0+23};
// lanes 16..31 -> same rows, K shifted by +8. Two b128 loads per lane.
static __device__ __forceinline__ v16bf load_a16x32(const __bf16* __restrict__ p,
                                                    int ld, int row0, int k0) {
  unsigned lane = laneid(), half = lane >> 4, m = lane & 15u;
  const __bf16* rp = p + (size_t)(row0 + (int)m) * (size_t)ld + k0 + (int)half * 8;
  v8bf lo = *(const v8bf*)(rp);
  v8bf hi = *(const v8bf*)(rp + 16);
  v16bf a;
#pragma unroll
  for (int i = 0; i < 8; ++i) { a[i] = lo[i]; a[i + 8] = hi[i]; }
  return a;
}

// B (32x16 bf16, KxN) where element (K,N) = mem[(n0+N)*ld + k0 + K]
// (each output column is a contiguous row in memory — rows of W, rows of
// K-tiles, rows of V^T). lane<16 -> N=lane, K in {0..7, 16..23}; lanes 16..31 -> K shifted by +8.
static __device__ __forceinline__ v16bf load_b32x16(const __bf16* __restrict__ p,
                                                    int ld, int n0, int k0) {
  unsigned lane = laneid(), half = lane >> 4, n = lane & 15u;
  const __bf16* rp = p + (size_t)(n0 + (int)n) * (size_t)ld + k0 + (int)half * 8;
  v8bf lo = *(const v8bf*)(rp);
  v8bf hi = *(const v8bf*)(rp + 16);
  v16bf b;
#pragma unroll
  for (int i = 0; i < 8; ++i) { b[i] = lo[i]; b[i + 8] = hi[i]; }
  return b;
}

static __device__ __forceinline__ v8f bf16_wmma(v16bf a, v16bf b, v8f c) {
  return __builtin_amdgcn_wmma_f32_16x16x32_bf16(false, a, false, b,
                                                 (short)0, c, false, false);
}

static __device__ __forceinline__ float rowmax16(float x) {
#pragma unroll
  for (int m = 1; m < 16; m <<= 1) x = fmaxf(x, __shfl_xor(x, m, 32));
  return x;
}
static __device__ __forceinline__ float rowsum16(float x) {
#pragma unroll
  for (int m = 1; m < 16; m <<= 1) x += __shfl_xor(x, m, 32);
  return x;
}

// ---- fp32 -> bf16 cast (vectorized) ----------------------------------------
__global__ __launch_bounds__(256)
void cvt_f32_to_bf16(const float* __restrict__ in, __bf16* __restrict__ out, int n4) {
  int i = blockIdx.x * blockDim.x + threadIdx.x;
  if (i >= n4) return;
  float4 f = ((const float4*)in)[i];
  v4bf o;
  o[0] = (__bf16)f.x; o[1] = (__bf16)f.y; o[2] = (__bf16)f.z; o[3] = (__bf16)f.w;
  ((v4bf*)out)[i] = o;
}

// ---- GEMM: Y = X @ W^T + bias. One wave -> 32x64 C tile (B frags reused 2x).
// MODE 0: store fp32 row-major (M x N)                 (output projection)
// MODE 1: store bf16 split-head (B,H,S,HS)             (Q,K projections)
// MODE 2: store bf16 split-head transposed (B,H,HS,S)  (V projection)
template <int MODE>
__global__ __launch_bounds__(256)
void gemm_bf16_wmma(const __bf16* __restrict__ A, const __bf16* __restrict__ W,
                    const float* __restrict__ bias, void* __restrict__ out,
                    int M, int N, int K) {
  unsigned wid = threadIdx.x >> 5;
  unsigned wg  = blockIdx.x * 8u + wid;
  unsigned ntiles = (unsigned)N / 64u;
  unsigned mt = wg / ntiles, nt = wg % ntiles;
  int m0 = (int)mt * 32, n0 = (int)nt * 64;
  if (m0 >= M) return;

  v8f c[2][4];
#pragma unroll
  for (int u = 0; u < 2; ++u)
#pragma unroll
    for (int t = 0; t < 4; ++t) { v8f z = {}; c[u][t] = z; }

  unsigned lane = laneid(), half = lane >> 4, nl = lane & 15u;

  for (int k0 = 0; k0 < K; k0 += 32) {
    v16bf a0 = load_a16x32(A, K, m0,      k0);
    v16bf a1 = load_a16x32(A, K, m0 + 16, k0);
    if (k0 + 128 < K) {   // speculative gfx1250 prefetch (global_prefetch_b8)
      __builtin_prefetch(A + (size_t)(m0 + (int)lane) * K + k0 + 128, 0, 1);
      __builtin_prefetch(W + (size_t)(n0 + (int)(lane & 31)) * K + k0 + 128, 0, 1);
      __builtin_prefetch(W + (size_t)(n0 + 32 + (int)(lane & 31)) * K + k0 + 128, 0, 1);
    }
#pragma unroll
    for (int t = 0; t < 4; ++t) {
      v16bf b = load_b32x16(W, K, n0 + 16 * t, k0);
      c[0][t] = bf16_wmma(a0, b, c[0][t]);
      c[1][t] = bf16_wmma(a1, b, c[1][t]);
    }
  }

#pragma unroll
  for (int u = 0; u < 2; ++u) {
#pragma unroll
    for (int t = 0; t < 4; ++t) {
      int n = n0 + 16 * t + (int)nl;
      float bv = bias[n];
#pragma unroll
      for (int r = 0; r < 8; ++r) {
        int m = m0 + 16 * u + r + 8 * (int)half;
        float v = c[u][t][r] + bv;
        if (MODE == 0) {
          ((float*)out)[(size_t)m * N + n] = v;
        } else {
          int b_ = m >> 11, s_ = m & (NS - 1);   // m -> (batch, seq)
          int h_ = n >> 6,  dl = n & (NHS - 1);  // n -> (head, head-dim)
          __bf16 bw = (__bf16)v;
          if (MODE == 1)
            ((__bf16*)out)[(((size_t)(b_ * NH + h_)) * NS + s_) * NHS + dl] = bw;
          else
            ((__bf16*)out)[(((size_t)(b_ * NH + h_)) * NHS + dl) * NS + s_] = bw;
        }
      }
    }
  }
}

// ---- Fused causal flash attention. One wave -> 16 query rows of one head. --
// Masking only on diagonal-overlapping key blocks (MASKED template = folded).
template <bool MASKED>
static __device__ __forceinline__ void attn_step(
    int key0, int m0, const __bf16* __restrict__ k, const __bf16* __restrict__ vt,
    __bf16* pl, v16bf a0, v16bf a1, v8f (&acc)[4], float (&mrow)[8],
    float (&lrow)[8], unsigned half, unsigned nl) {
  v8f s0 = {}, s1 = {};
  {
    v16bf b00 = load_b32x16(k, NHS, key0,      0);
    v16bf b01 = load_b32x16(k, NHS, key0,      32);
    s0 = bf16_wmma(a0, b00, s0);
    s0 = bf16_wmma(a1, b01, s0);
    v16bf b10 = load_b32x16(k, NHS, key0 + 16, 0);
    v16bf b11 = load_b32x16(k, NHS, key0 + 16, 32);
    s1 = bf16_wmma(a0, b10, s1);
    s1 = bf16_wmma(a1, b11, s1);
  }
  if (key0 + 32 <= m0 + 15)   // prefetch next key block (K rows, 1B/lane-row)
    __builtin_prefetch(k + (size_t)(key0 + 32 + (int)(half * 16u + nl)) * NHS, 0, 1);
#pragma unroll
  for (int r = 0; r < 8; ++r) {
    float x0 = s0[r] * ATTN_SCALE;
    float x1 = s1[r] * ATTN_SCALE;
    if (MASKED) {
      int row = m0 + r + 8 * (int)half;
      if (key0 + (int)nl > row)      x0 = -1.0e9f;   // causal mask
      if (key0 + 16 + (int)nl > row) x1 = -1.0e9f;
    }
    float mx   = rowmax16(fmaxf(x0, x1));
    float mnew = fmaxf(mrow[r], mx);
    float al   = __expf(mrow[r] - mnew);
    float p0   = __expf(x0 - mnew);
    float p1   = __expf(x1 - mnew);
    lrow[r] = lrow[r] * al + rowsum16(p0 + p1);
    mrow[r] = mnew;
#pragma unroll
    for (int t = 0; t < 4; ++t) acc[t][r] = acc[t][r] * al;   // rescale O
    int prow = r + 8 * (int)half;
    pl[prow * 32 + (int)nl]      = (__bf16)p0;   // P tile, row-major 16x32
    pl[prow * 32 + 16 + (int)nl] = (__bf16)p1;
  }
  // C-layout -> A-layout transpose through LDS; DS is in-order per wave,
  // explicit CDNA5 split-counter wait keeps the compiler honest too.
  asm volatile("s_wait_dscnt 0" ::: "memory");
  v16bf pa = load_a16x32(pl, 32, 0, 0);
  // O(16x64) += P(16x32) @ V(32x64); V^T rows are contiguous along keys
#pragma unroll
  for (int t = 0; t < 4; ++t) {
    v16bf vb = load_b32x16(vt, NS, 16 * t, key0);
    acc[t] = bf16_wmma(pa, vb, acc[t]);
  }
}

// Q,K: (B,H,S,HS) bf16.  VT: (B,H,HS,S) bf16.  O: (B,H,S,HS) bf16.
__global__ __launch_bounds__(256)
void attn_fused_wmma(const __bf16* __restrict__ Q, const __bf16* __restrict__ Kh,
                     const __bf16* __restrict__ VT, __bf16* __restrict__ O) {
  __shared__ __bf16 pbuf[8][16 * 32];   // per-wave P-tile staging
  unsigned wid = threadIdx.x >> 5;
  unsigned wg  = blockIdx.x * 8u + wid;
  unsigned bh  = wg >> 7;               // (batch*head) : 0..31
  unsigned qt  = wg & 127u;             // query tile within head
  int m0 = (int)qt * 16;

  const __bf16* q  = Q  + (size_t)bh * NS * NHS;
  const __bf16* k  = Kh + (size_t)bh * NS * NHS;
  const __bf16* vt = VT + (size_t)bh * NHS * NS;
  __bf16*       o  = O  + (size_t)bh * NS * NHS;
  __bf16*       pl = pbuf[wid];

  unsigned lane = laneid(), half = lane >> 4, nl = lane & 15u;

  v16bf a0 = load_a16x32(q, NHS, m0, 0);    // q rows, d 0..31
  v16bf a1 = load_a16x32(q, NHS, m0, 32);   // q rows, d 32..63

  v8f acc[4];
#pragma unroll
  for (int t = 0; t < 4; ++t) { v8f z = {}; acc[t] = z; }
  float mrow[8], lrow[8];
#pragma unroll
  for (int r = 0; r < 8; ++r) { mrow[r] = -3.0e38f; lrow[r] = 0.0f; }

  // blocks with key0+31 <= m0 never touch the diagonal -> no masking needed
  int fullEnd = m0 & ~31;
  for (int key0 = 0; key0 < fullEnd; key0 += 32)
    attn_step<false>(key0, m0, k, vt, pl, a0, a1, acc, mrow, lrow, half, nl);
  for (int key0 = fullEnd; key0 <= m0 + 15; key0 += 32)
    attn_step<true>(key0, m0, k, vt, pl, a0, a1, acc, mrow, lrow, half, nl);

  // normalize and store attended in (B,H,S,HS) — flat view == reference reshape
#pragma unroll
  for (int r = 0; r < 8; ++r) {
    float inv = 1.0f / lrow[r];
    int row = m0 + r + 8 * (int)half;
#pragma unroll
    for (int t = 0; t < 4; ++t)
      o[(size_t)row * NHS + 16 * t + (int)nl] = (__bf16)(acc[t][r] * inv);
  }
}

// ---- host-side orchestration ------------------------------------------------
extern "C" void kernel_launch(void* const* d_in, const int* in_sizes, int n_in,
                              void* d_out, int out_size, void* d_ws, size_t ws_size,
                              hipStream_t stream) {
  const float* Xq = (const float*)d_in[0];
  const float* Xk = (const float*)d_in[1];
  const float* Xv = (const float*)d_in[2];
  const float* Wq = (const float*)d_in[3];
  const float* bq = (const float*)d_in[4];
  const float* Wk = (const float*)d_in[5];
  const float* bk = (const float*)d_in[6];
  const float* Wv = (const float*)d_in[7];
  const float* bv = (const float*)d_in[8];
  const float* Wo = (const float*)d_in[9];
  const float* bo = (const float*)d_in[10];

  const size_t MB = 1024 * 1024;
  char* ws = (char*)d_ws;
  __bf16* xq = (__bf16*)(ws + 0 * MB);    // 8 MB each
  __bf16* xk = (__bf16*)(ws + 8 * MB);
  __bf16* xv = (__bf16*)(ws + 16 * MB);
  __bf16* wq = (__bf16*)(ws + 24 * MB);   // 2 MB each
  __bf16* wk = (__bf16*)(ws + 26 * MB);
  __bf16* wv = (__bf16*)(ws + 28 * MB);
  __bf16* wo = (__bf16*)(ws + 30 * MB);
  __bf16* Qh = (__bf16*)(ws + 32 * MB);   // 8 MB each
  __bf16* Kh = (__bf16*)(ws + 40 * MB);
  __bf16* VT = (__bf16*)(ws + 48 * MB);
  __bf16* AT = (__bf16*)(ws + 56 * MB);   // 64 MB total (fits 192 MB L2)

  const int nX = NB * NS * ND;  // 4194304
  const int nW = ND * ND;       // 1048576
  const int M = NB * NS, N = ND, K = ND;

  cvt_f32_to_bf16<<<(nX / 4 + 255) / 256, 256, 0, stream>>>(Xq, xq, nX / 4);
  cvt_f32_to_bf16<<<(nX / 4 + 255) / 256, 256, 0, stream>>>(Xk, xk, nX / 4);
  cvt_f32_to_bf16<<<(nX / 4 + 255) / 256, 256, 0, stream>>>(Xv, xv, nX / 4);
  cvt_f32_to_bf16<<<(nW / 4 + 255) / 256, 256, 0, stream>>>(Wq, wq, nW / 4);
  cvt_f32_to_bf16<<<(nW / 4 + 255) / 256, 256, 0, stream>>>(Wk, wk, nW / 4);
  cvt_f32_to_bf16<<<(nW / 4 + 255) / 256, 256, 0, stream>>>(Wv, wv, nW / 4);
  cvt_f32_to_bf16<<<(nW / 4 + 255) / 256, 256, 0, stream>>>(Wo, wo, nW / 4);

  // (M/32)*(N/64) = 2048 waves -> 256 blocks of 8 waves
  const int gemmBlocks = (M / 32) * (N / 64) / 8;
  gemm_bf16_wmma<1><<<gemmBlocks, 256, 0, stream>>>(xq, wq, bq, Qh, M, N, K);
  gemm_bf16_wmma<1><<<gemmBlocks, 256, 0, stream>>>(xk, wk, bk, Kh, M, N, K);
  gemm_bf16_wmma<2><<<gemmBlocks, 256, 0, stream>>>(xv, wv, bv, VT, M, N, K);

  // B*H*(S/16) = 4096 waves -> 512 blocks
  attn_fused_wmma<<<NB * NH * (NS / 16) / 8, 256, 0, stream>>>(Qh, Kh, VT, AT);

  // flat (B,H,S,HS) view of AT == reference's contiguous reshape to (B,S,D)
  gemm_bf16_wmma<0><<<gemmBlocks, 256, 0, stream>>>(AT, wo, bo, d_out, M, N, K);
}
